// CausalAttention_33930241639029
// MI455X (gfx1250) — matrix operand order repaired and checked by reading, not compiled, not run on hardware
//
#include <hip/hip_runtime.h>
#include <hip/hip_bf16.h>

// ---------------------------------------------------------------------------
// CDNA5 (gfx1250, wave32) implementation of the causal-attention + delta-rule
// memory reference.  All matmuls via v_wmma_f32_16x16x32_f16; tile staging via
// GLOBAL_LOAD_ASYNC_TO_LDS_B128 (ASYNCcnt) where contiguous.
// ---------------------------------------------------------------------------

typedef __attribute__((ext_vector_type(16))) _Float16 v16h;
typedef __attribute__((ext_vector_type(8)))  float    v8f;
typedef __attribute__((ext_vector_type(4)))  int      v4i;

#define DEV __device__ __forceinline__

static constexpr int BATCH = 2;
static constexpr int NPOS  = 2048;
static constexpr int NDIM  = 1024;
static constexpr int HEADS = 8;
static constexpr int DH    = 128;
static constexpr int BHN   = BATCH * HEADS;      // 16
static constexpr int ROWS  = BATCH * NPOS;       // 4096
static constexpr float SCALE = 0.08838834764831845f;   // 128^-0.5
static constexpr float EPSV  = 1e-10f;

union AFrag { v16h v; _Float16 h[16]; };

#if __has_builtin(__builtin_amdgcn_global_load_async_to_lds_b128)
#define HAVE_ASYNC_LDS 1
#else
#define HAVE_ASYNC_LDS 0
#endif

typedef __attribute__((address_space(1))) v4i* gptr_v4i;
typedef __attribute__((address_space(3))) v4i* lptr_v4i;

// copy 16 contiguous bytes global -> LDS (async path on gfx1250)
DEV void async_copy16(const _Float16* g, _Float16* l) {
#if HAVE_ASYNC_LDS
    __builtin_amdgcn_global_load_async_to_lds_b128(
        (gptr_v4i)(v4i*)g, (lptr_v4i)(v4i*)l, 0, 0);
#else
    *(uint4*)l = *(const uint4*)g;
#endif
}

DEV void wait_async_lds() {
#if HAVE_ASYNC_LDS
#if __has_builtin(__builtin_amdgcn_s_wait_asynccnt)
    __builtin_amdgcn_s_wait_asynccnt(0);
#else
    asm volatile("s_wait_asynccnt 0x0" ::: "memory");
#endif
#endif
}

// interleaved-K index for the 16-bit A fragment (ISA 7.12.2)
DEV int a_kbase(int v, int hi) {
    int base = (v < 4) ? (2 * v) : (16 + 2 * (v - 4));
    return base + (hi ? 8 : 0);
}

// A fragment (16x32 f16) from an LDS tile laid out row-major [16+][ldk]
DEV v16h load_a_lds(const _Float16* base, int ldk) {
    int l = threadIdx.x & 31;
    int m = l & 15, hi = l >> 4;
    AFrag f;
#pragma unroll
    for (int v = 0; v < 8; ++v) {
        int kb = a_kbase(v, hi);
        f.h[2 * v]     = base[m * ldk + kb];
        f.h[2 * v + 1] = base[m * ldk + kb + 1];
    }
    return f.v;
}

// B fragment (32x16 f16): lane = column, elements = interleaved-K contraction rows.
DEV v16h load_b_lds(const _Float16* base, int ldn, int c0) {
    int l = threadIdx.x & 31;
    int n = l & 15, hi = l >> 4;
    AFrag f;
#pragma unroll
    for (int v = 0; v < 8; ++v) {
        int kb = a_kbase(v, hi);
        f.h[2 * v]     = base[kb * ldn + c0 + n];
        f.h[2 * v + 1] = base[(kb + 1) * ldn + c0 + n];
    }
    return f.v;
}

DEV v8f wmma_f16(v16h a, v16h b, v8f c) {
    return __builtin_amdgcn_wmma_f32_16x16x32_f16(false, a, false, b, (short)0, c,
                                                  false, false);
}

// ---------------------------------------------------------------------------
// f32 -> f16 convert
// ---------------------------------------------------------------------------
__global__ void k_convert_f16(const float* __restrict__ in, _Float16* __restrict__ out,
                              int n) {
    int i = blockIdx.x * blockDim.x + threadIdx.x;
    if (i < n) out[i] = (_Float16)in[i];
}

// ---------------------------------------------------------------------------
// RMSNorm: xn = x / max(||x||,1e-12) * sqrt(DIM) * gamma  (one block per row)
// ---------------------------------------------------------------------------
__global__ void k_rmsnorm(const float* __restrict__ x, const float* __restrict__ gamma,
                          _Float16* __restrict__ xn) {
    __shared__ float red[256];
    int row = blockIdx.x, tid = threadIdx.x;
    const float* xr = x + (size_t)row * NDIM;
    float ss = 0.f;
#pragma unroll
    for (int i = 0; i < 4; ++i) { float v = xr[tid + 256 * i]; ss += v * v; }
    red[tid] = ss; __syncthreads();
    for (int s = 128; s > 0; s >>= 1) {
        if (tid < s) red[tid] += red[tid + s];
        __syncthreads();
    }
    float scale = 32.0f / fmaxf(sqrtf(red[0]), 1e-12f);
#pragma unroll
    for (int i = 0; i < 4; ++i) {
        int c = tid + 256 * i;
        xn[(size_t)row * NDIM + c] = (_Float16)(xr[c] * scale * gamma[c]);
    }
}

// ---------------------------------------------------------------------------
// Generic batched WMMA GEMM:  C[m,n] = sum_k A'[m,k]*B'[k,n] (+ Cin)
//   A'[m,k] = At ? A[k*lda+m] : A[m*lda+k]
//   B'[k,n] = Bt ? B[n*ldb+k] : B[k*ldb+n]
// Block: 256 threads (8 waves), tile 128x128, k-chunk 32.
// Contiguous (non-transposed) tiles staged with async b128 LDS loads.
// ---------------------------------------------------------------------------
__global__ void __launch_bounds__(256, 1)
k_gemm_wmma(const _Float16* __restrict__ A, const _Float16* __restrict__ B,
            const float* __restrict__ Cin, float* __restrict__ Cout,
            int M, int N, int K, int lda, int ldb, int ldc,
            int At, int Bt,
            long long sA, long long sB, long long sCin, long long sC) {
    __shared__ __align__(16) _Float16 As[128][40];
    __shared__ __align__(16) _Float16 Bs[32][136];

    int z = blockIdx.z;
    const _Float16* Ab = A + z * sA;
    const _Float16* Bb = B + z * sB;
    int row0 = blockIdx.y * 128, col0 = blockIdx.x * 128;
    int tid = threadIdx.x, w = tid >> 5, lane = tid & 31;
    int m = lane & 15, hi = lane >> 4;

    v8f acc[8];
#pragma unroll
    for (int t = 0; t < 8; ++t) acc[t] = (v8f)0.f;

    for (int k0 = 0; k0 < K; k0 += 32) {
        bool used_async = false;
        // ---- stage A tile 128x32 ----
        if (!At) {          // contiguous 16B chunks along k
#pragma unroll
            for (int i = 0; i < 2; ++i) {
                int c = tid + i * 256;          // 0..511
                int r = c >> 2;                 // 4 chunks per row
                int cc = (c & 3) * 8;           // half offset in row
                if (row0 + r < M && k0 + cc + 8 <= K)
                    async_copy16(&Ab[(size_t)(row0 + r) * lda + k0 + cc], &As[r][cc]);
            }
            used_async = true;
        } else {
#pragma unroll
            for (int i = 0; i < 16; ++i) {
                int e = tid + i * 256;
                int r = e >> 5, kk = e & 31;
                _Float16 v = (_Float16)0.f;
                if (row0 + r < M && k0 + kk < K)
                    v = Ab[(size_t)(k0 + kk) * lda + row0 + r];
                As[r][kk] = v;
            }
        }
        // ---- stage B tile 32x128 ----
        if (!Bt) {          // contiguous 16B chunks along n
#pragma unroll
            for (int i = 0; i < 2; ++i) {
                int c = tid + i * 256;
                int r = c >> 4;                 // 16 chunks per row
                int cc = (c & 15) * 8;
                if (k0 + r < K && col0 + cc + 8 <= N)
                    async_copy16(&Bb[(size_t)(k0 + r) * ldb + col0 + cc], &Bs[r][cc]);
            }
            used_async = true;
        } else {
#pragma unroll
            for (int i = 0; i < 16; ++i) {
                int e = tid + i * 256;
                int kk = e >> 7, c = e & 127;
                _Float16 v = (_Float16)0.f;
                if (col0 + c < N && k0 + kk < K)
                    v = Bb[(size_t)(col0 + c) * ldb + k0 + kk];
                Bs[kk][c] = v;
            }
        }
        if (used_async) wait_async_lds();
        __syncthreads();

        if (k0 + 32 < K)
            __builtin_prefetch(&Ab[At ? (size_t)(k0 + 32) * lda + row0
                                      : (size_t)row0 * lda + k0 + 32], 0, 1);
        v16h a = load_a_lds(&As[w * 16][0], 40);
#pragma unroll
        for (int t = 0; t < 8; ++t) {
            v16h b = load_b_lds(&Bs[0][0], 136, t * 16);
            acc[t] = wmma_f16(a, b, acc[t]);
        }
        __syncthreads();
    }

    float* Co = Cout + z * sC;
    const float* Ci = Cin ? Cin + z * sCin : nullptr;
#pragma unroll
    for (int t = 0; t < 8; ++t) {
#pragma unroll
        for (int r = 0; r < 8; ++r) {
            int row = row0 + w * 16 + r + 8 * hi;
            int col = col0 + t * 16 + m;
            if (row < M && col < N) {
                float v = acc[t][r];
                if (Ci) v += Ci[(size_t)row * ldc + col];
                Co[(size_t)row * ldc + col] = v;
            }
        }
    }
}

// ---------------------------------------------------------------------------
// RoPE + feature-map / layout split.
// qkv f32 [b, n, 3, h, dh]  ->  per-(b,h,n) q_rot/k_rot/v f16, v f32, qf/kf f16
// ---------------------------------------------------------------------------
__global__ void k_rope_feat(const float* __restrict__ qkv,
                            _Float16* __restrict__ qrot, _Float16* __restrict__ krot,
                            _Float16* __restrict__ v16, float* __restrict__ v32,
                            _Float16* __restrict__ qf, _Float16* __restrict__ kf) {
    int idx = blockIdx.x * 256 + threadIdx.x;              // 2^21 total
    int i    = idx & 63;
    int h    = (idx >> 6) & 7;
    int npos = (idx >> 9) & 2047;
    int b    = idx >> 20;
    int bh   = b * HEADS + h;

    size_t base = ((size_t)(b * NPOS + npos)) * (3 * NDIM) + h * DH + 2 * i;
    float q1 = qkv[base],             q2 = qkv[base + 1];
    float k1 = qkv[base + NDIM],      k2 = qkv[base + NDIM + 1];
    float vv1 = qkv[base + 2 * NDIM], vv2 = qkv[base + 2 * NDIM + 1];

    float inv_freq = __powf(10000.0f, -(float)(2 * i) / (float)DH);
    float ang = (float)npos * inv_freq;
    float c = __cosf(ang), s = __sinf(ang);

    size_t o = ((size_t)bh * NPOS + npos) * DH + 2 * i;
    qrot[o]     = (_Float16)(SCALE * (q1 * c - q2 * s));
    qrot[o + 1] = (_Float16)(SCALE * (q2 * c + q1 * s));
    krot[o]     = (_Float16)(k1 * c - k2 * s);
    krot[o + 1] = (_Float16)(k2 * c + k1 * s);
    v16[o] = (_Float16)vv1;  v16[o + 1] = (_Float16)vv2;
    v32[o] = vv1;            v32[o + 1] = vv2;
    // elu(x)+1  ==  x>0 ? x+1 : exp(x)
    qf[o]     = (_Float16)(q1 > 0.f ? q1 + 1.f : __expf(q1));
    qf[o + 1] = (_Float16)(q2 > 0.f ? q2 + 1.f : __expf(q2));
    kf[o]     = (_Float16)(k1 > 0.f ? k1 + 1.f : __expf(k1));
    kf[o + 1] = (_Float16)(k2 > 0.f ? k2 + 1.f : __expf(k2));
}

// ---------------------------------------------------------------------------
// Causal flash attention.  grid (n/64, b*h), block 128 (4 waves).
// Wave w owns 16 query rows.  32-key blocks; K/V async-staged into LDS;
// P staged in LDS to convert C-layout -> A-fragment layout.
// ---------------------------------------------------------------------------
__global__ void __launch_bounds__(128, 1)
k_flash_attn(const _Float16* __restrict__ Q,
             const _Float16* __restrict__ Kp,
             const _Float16* __restrict__ V,
             float* __restrict__ O) {
    __shared__ __align__(16) _Float16 Ks[32][136];
    __shared__ __align__(16) _Float16 Vs[32][136];
    __shared__ __align__(16) _Float16 Ps[4][16][36];

    int bh = blockIdx.y;
    int tid = threadIdx.x, w = tid >> 5, lane = tid & 31;
    int m = lane & 15, hi = lane >> 4;
    int q0 = blockIdx.x * 64 + w * 16;                 // global q row in [0,n)

    const _Float16* Qb = Q + (size_t)bh * NPOS * DH;
    const _Float16* Kb = Kp + (size_t)bh * NPOS * DH;
    const _Float16* Vb = V + (size_t)bh * NPOS * DH;

    // resident Q fragments (16 rows x 128 d, 4 chunks of 32)
    v16h qa[4];
#pragma unroll
    for (int c = 0; c < 4; ++c) {
        AFrag f;
#pragma unroll
        for (int v = 0; v < 8; ++v) {
            int kb = a_kbase(v, hi);
            f.h[2 * v]     = Qb[(size_t)(q0 + m) * DH + c * 32 + kb];
            f.h[2 * v + 1] = Qb[(size_t)(q0 + m) * DH + c * 32 + kb + 1];
        }
        qa[c] = f.v;
    }

    v8f acc[8];
#pragma unroll
    for (int t = 0; t < 8; ++t) acc[t] = (v8f)0.f;
    float mi[8], li[8];
#pragma unroll
    for (int r = 0; r < 8; ++r) { mi[r] = -1e30f; li[r] = 0.f; }

    int nkb = blockIdx.x * 2 + 2;                      // same for all waves
    for (int kb = 0; kb < nkb; ++kb) {
        int kk0 = kb * 32;
        // async stage K/V 32x128 (16B chunks; 512 chunks per buffer, 4/thread)
#pragma unroll
        for (int i = 0; i < 4; ++i) {
            int c = tid + i * 128;
            int r = c >> 4, cc = (c & 15) * 8;
            async_copy16(&Kb[(size_t)(kk0 + r) * DH + cc], &Ks[r][cc]);
            async_copy16(&Vb[(size_t)(kk0 + r) * DH + cc], &Vs[r][cc]);
        }
        wait_async_lds();
        __syncthreads();

        // S = Q K^T  (two 16-col tiles)
        v8f s[2]; s[0] = (v8f)0.f; s[1] = (v8f)0.f;
#pragma unroll
        for (int c = 0; c < 4; ++c) {
#pragma unroll
            for (int t = 0; t < 2; ++t) {
                AFrag bf;
#pragma unroll
                for (int j = 0; j < 16; ++j)
                    bf.h[j] = Ks[t * 16 + m][c * 32 + a_kbase(j >> 1, hi) + (j & 1)];   // B row = d, col = key
                s[t] = wmma_f16(qa[c], bf.v, s[t]);
            }
        }

        // causal mask + online softmax
#pragma unroll
        for (int r = 0; r < 8; ++r) {
            int qrow = q0 + r + 8 * hi;
            float s0 = s[0][r], s1 = s[1][r];
            if (kk0 + m > qrow)      s0 = -1e30f;
            if (kk0 + 16 + m > qrow) s1 = -1e30f;
            float rmax = fmaxf(s0, s1);
#pragma unroll
            for (int off = 1; off < 16; off <<= 1)
                rmax = fmaxf(rmax, __shfl_xor(rmax, off, 32));
            float mnew = fmaxf(mi[r], rmax);
            float resc = __expf(mi[r] - mnew);
            float p0 = __expf(s0 - mnew);
            float p1 = __expf(s1 - mnew);
            float rsum = p0 + p1;
#pragma unroll
            for (int off = 1; off < 16; off <<= 1)
                rsum += __shfl_xor(rsum, off, 32);
            li[r] = li[r] * resc + rsum;
            mi[r] = mnew;
#pragma unroll
            for (int t = 0; t < 8; ++t) acc[t][r] *= resc;
            Ps[w][r + 8 * hi][m]      = (_Float16)p0;
            Ps[w][r + 8 * hi][16 + m] = (_Float16)p1;
        }
        __syncthreads();

        // O += P V
        v16h pa = load_a_lds(&Ps[w][0][0], 36);
#pragma unroll
        for (int t = 0; t < 8; ++t) {
            v16h bf = load_b_lds(&Vs[0][0], 136, t * 16);
            acc[t] = wmma_f16(pa, bf, acc[t]);
        }
        __syncthreads();
    }

    float* Ob = O + (size_t)bh * NPOS * DH;
#pragma unroll
    for (int t = 0; t < 8; ++t)
#pragma unroll
        for (int r = 0; r < 8; ++r) {
            int row = q0 + r + 8 * hi;
            Ob[(size_t)row * DH + t * 16 + m] = acc[t][r] / fmaxf(li[r], 1e-30f);
        }
}

// ---------------------------------------------------------------------------
// Gated merge: merged[b,n,h*dh] = attn*g + (numer/clip(qf.mem_norm))*(1-g)
// ---------------------------------------------------------------------------
__global__ void k_merge_gate(const float* __restrict__ attn, const float* __restrict__ numer,
                             const _Float16* __restrict__ qf,
                             const float* __restrict__ mem_norm,
                             const float* __restrict__ head_gates,
                             _Float16* __restrict__ merged) {
    __shared__ float red[128];
    int npos = blockIdx.x & 2047;
    int bh = blockIdx.x >> 11;
    int h = bh & 7, b = bh >> 3;
    int dd = threadIdx.x;
    size_t row = (size_t)bh * NPOS + npos;
    red[dd] = (float)qf[row * DH + dd] * mem_norm[(size_t)bh * DH + dd];
    __syncthreads();
    for (int s = 64; s > 0; s >>= 1) {
        if (dd < s) red[dd] += red[dd + s];
        __syncthreads();
    }
    float denom = fmaxf(red[0], EPSV);
    float g = 1.f / (1.f + __expf(-head_gates[h]));
    float mo = numer[row * DH + dd] / denom;
    float val = attn[row * DH + dd] * g + mo * (1.f - g);
    merged[((size_t)(b * NPOS + npos)) * NDIM + h * DH + dd] = (_Float16)val;
}

// ---------------------------------------------------------------------------
// v_new = v - numer/clip(kf.mem_norm)   -> f16
// ---------------------------------------------------------------------------
__global__ void k_vnew(const float* __restrict__ v32, const float* __restrict__ numer,
                       const _Float16* __restrict__ kf,
                       const float* __restrict__ mem_norm,
                       _Float16* __restrict__ vnew) {
    __shared__ float red[128];
    int npos = blockIdx.x & 2047;
    int bh = blockIdx.x >> 11;
    int dd = threadIdx.x;
    size_t row = (size_t)bh * NPOS + npos;
    red[dd] = (float)kf[row * DH + dd] * mem_norm[(size_t)bh * DH + dd];
    __syncthreads();
    for (int s = 64; s > 0; s >>= 1) {
        if (dd < s) red[dd] += red[dd + s];
        __syncthreads();
    }
    float denom = fmaxf(red[0], EPSV);
    vnew[row * DH + dd] = (_Float16)(v32[row * DH + dd] - numer[row * DH + dd] / denom);
}

// ---------------------------------------------------------------------------
// new_norm = sum_n kf + mem_norm ;  16 blocks x 128 threads
// ---------------------------------------------------------------------------
__global__ void k_newnorm(const _Float16* __restrict__ kf,
                          const float* __restrict__ mem_norm,
                          float* __restrict__ out) {
    int bh = blockIdx.x, kd = threadIdx.x;
    const _Float16* kb = kf + (size_t)bh * NPOS * DH;
    float acc = mem_norm[(size_t)bh * DH + kd];
    for (int nn = 0; nn < NPOS; ++nn) acc += (float)kb[(size_t)nn * DH + kd];
    out[(size_t)bh * DH + kd] = acc;
}

// ---------------------------------------------------------------------------
// host launcher
// ---------------------------------------------------------------------------
extern "C" void kernel_launch(void* const* d_in, const int* in_sizes, int n_in,
                              void* d_out, int out_size, void* d_ws, size_t ws_size,
                              hipStream_t stream) {
    const float* x          = (const float*)d_in[0];
    const float* gamma      = (const float*)d_in[1];
    const float* w_qkv      = (const float*)d_in[2];
    const float* w_out      = (const float*)d_in[3];
    const float* head_gates = (const float*)d_in[4];
    const float* mem_kv     = (const float*)d_in[5];
    const float* mem_norm   = (const float*)d_in[6];
    float* out = (float*)d_out;

    float* out_main = out;                              // [b,n,1024]
    float* out_kv   = out + (size_t)ROWS * NDIM;        // [b,h,128,128]
    float* out_norm = out_kv + (size_t)BHN * DH * DH;   // [b,h,128]

    char* ws = (char*)d_ws;
    size_t off = 0;
    auto alloc = [&](size_t bytes) {
        size_t o = off; off = (off + bytes + 255) & ~(size_t)255; return o;
    };
    _Float16* wqkv16  = (_Float16*)(ws + alloc((size_t)3 * NDIM * NDIM * 2)); // 3072x1024
    _Float16* wout16  = (_Float16*)(ws + alloc((size_t)NDIM * NDIM * 2));
    _Float16* mkv16   = (_Float16*)(ws + alloc((size_t)BHN * DH * DH * 2));
    _Float16* xn16    = (_Float16*)(ws + alloc((size_t)ROWS * NDIM * 2));
    float*    qkv32   = (float*)   (ws + alloc((size_t)ROWS * 3 * NDIM * 4));
    float*    v32     = (float*)   (ws + alloc((size_t)BHN * NPOS * DH * 4));
    _Float16* qrot16  = (_Float16*)(ws + alloc((size_t)BHN * NPOS * DH * 2));
    _Float16* krot16  = (_Float16*)(ws + alloc((size_t)BHN * NPOS * DH * 2));
    _Float16* v16     = (_Float16*)(ws + alloc((size_t)BHN * NPOS * DH * 2));
    _Float16* qf16    = (_Float16*)(ws + alloc((size_t)BHN * NPOS * DH * 2));
    _Float16* kf16    = (_Float16*)(ws + alloc((size_t)BHN * NPOS * DH * 2));
    _Float16* merged  = (_Float16*)(ws + alloc((size_t)ROWS * NDIM * 2));
    _Float16* vnew16  = (_Float16*)(ws + alloc((size_t)BHN * NPOS * DH * 2));
    // attn / numer alias the qkv32 slab (qkv32 fully consumed by k_rope_feat)
    float* attn32  = qkv32;
    float* numer32 = qkv32 + (size_t)BHN * NPOS * DH;

    const long long sQF = (long long)NPOS * DH;   // per-(b,h) feature stride
    const long long sKV = (long long)DH * DH;     // per-(b,h) mem_kv stride

    // 1-3. weight / memory converts
    k_convert_f16<<<(3 * NDIM * NDIM + 255) / 256, 256, 0, stream>>>(w_qkv, wqkv16,
                                                                     3 * NDIM * NDIM);
    k_convert_f16<<<(NDIM * NDIM + 255) / 256, 256, 0, stream>>>(w_out, wout16, NDIM * NDIM);
    k_convert_f16<<<(BHN * DH * DH + 255) / 256, 256, 0, stream>>>(mem_kv, mkv16,
                                                                   BHN * DH * DH);

    // 4. RMSNorm
    k_rmsnorm<<<ROWS, 256, 0, stream>>>(x, gamma, xn16);

    // 5. QKV projection: [4096x1024] x w_qkv^T -> qkv32 [4096x3072]
    {
        dim3 g(3 * NDIM / 128, ROWS / 128, 1);
        k_gemm_wmma<<<g, 256, 0, stream>>>(xn16, wqkv16, nullptr, qkv32,
                                           ROWS, 3 * NDIM, NDIM, NDIM, NDIM, 3 * NDIM,
                                           0, 1, 0, 0, 0, 0);
    }

    // 6. RoPE + features
    k_rope_feat<<<(BATCH * NPOS * HEADS * 64) / 256, 256, 0, stream>>>(
        qkv32, qrot16, krot16, v16, v32, qf16, kf16);

    // 7. flash attention -> attn32 [bh, n, dh]
    {
        dim3 g(NPOS / 64, BHN, 1);
        k_flash_attn<<<g, 128, 0, stream>>>(qrot16, krot16, v16, attn32);
    }

    // 8. retrieval numerator for qf: batched [2048x128]x[128x128]
    {
        dim3 g(1, NPOS / 128, BHN);
        k_gemm_wmma<<<g, 256, 0, stream>>>(qf16, mkv16, nullptr, numer32,
                                           NPOS, DH, DH, DH, DH, DH,
                                           0, 0, sQF, sKV, 0, sQF);
    }

    // 9. gated merge -> merged f16 [b,n,1024]
    k_merge_gate<<<BHN * NPOS, 128, 0, stream>>>(attn32, numer32, qf16, mem_norm,
                                                 head_gates, merged);

    // 10. out projection -> out_main
    {
        dim3 g(NDIM / 128, ROWS / 128, 1);
        k_gemm_wmma<<<g, 256, 0, stream>>>(merged, wout16, nullptr, out_main,
                                           ROWS, NDIM, NDIM, NDIM, NDIM, NDIM,
                                           0, 1, 0, 0, 0, 0);
    }

    // 11. retrieval numerator for kf (reuse numer32)
    {
        dim3 g(1, NPOS / 128, BHN);
        k_gemm_wmma<<<g, 256, 0, stream>>>(kf16, mkv16, nullptr, numer32,
                                           NPOS, DH, DH, DH, DH, DH,
                                           0, 0, sQF, sKV, 0, sQF);
    }

    // 12. v_new
    k_vnew<<<BHN * NPOS, 128, 0, stream>>>(v32, numer32, kf16, mem_norm, vnew16);

    // 13. new_kv = kf^T @ v_new + mem_kv  (batched 128x2048x128)
    {
        dim3 g(1, 1, BHN);
        k_gemm_wmma<<<g, 256, 0, stream>>>(kf16, vnew16, mem_kv, out_kv,
                                           DH, DH, NPOS, DH, DH, DH,
                                           1, 0, sQF, sQF, sKV, sKV);
    }

    // 14. new_norm
    k_newnorm<<<BHN, 128, 0, stream>>>(kf16, mem_norm, out_norm);

    (void)in_sizes; (void)n_in; (void)out_size; (void)ws_size;
}
